// LunaBlock_41455024341713
// MI455X (gfx1250) — matrix operand order repaired and checked by reading, not compiled, not run on hardware
//
#include <hip/hip_runtime.h>

// ---------------------------------------------------------------------------
// Luna cross-attention block for MI455X (gfx1250, wave32, WMMA, async LDS).
//
// - All GEMM operands are bf16 (weights pre-transposed to [N,K]; activations
//   produced as bf16 by GEMM/softmax/LN epilogues). fp32 kept only for
//   softmax logits, LayerNorm inputs and final outputs.
// - GEMM: 128x64 block tile, 8 waves, 32x32 wave tile (4 WMMAs/K-step),
//   double-buffered LDS fed by GLOBAL_LOAD_ASYNC_TO_LDS_B128 (ASYNCcnt).
// Workspace: ~650 MB.
// ---------------------------------------------------------------------------

typedef __attribute__((ext_vector_type(16))) __bf16 v16bf;
typedef __attribute__((ext_vector_type(8)))  __bf16 v8bf;
typedef __attribute__((ext_vector_type(8)))  float  v8f;

__device__ __forceinline__ __bf16 cvt_bf16(float f) {
    union { float f; unsigned u; } in; in.f = f;
    unsigned r = in.u + 0x7FFFu + ((in.u >> 16) & 1u);   // RNE
    union { unsigned short s; __bf16 b; } out;
    out.s = (unsigned short)(r >> 16);
    return out.b;
}

// async global -> LDS copy, 16B per lane, tracked by ASYNCcnt
__device__ __forceinline__ void async_b128(unsigned lds_off, const __bf16* g) {
    unsigned long long ga = (unsigned long long)(uintptr_t)g;
    asm volatile("global_load_async_to_lds_b128 %0, %1, off"
                 :: "v"(lds_off), "v"(ga) : "memory");
}
__device__ __forceinline__ void wait_async0() {
    asm volatile("s_wait_asynccnt 0x0" ::: "memory");
}

#define LDSP 40   // bf16 elems per LDS row (80B, multiple of 16B)

__device__ __forceinline__ v16bf frag_a(const __bf16* row, int halfSel) {
    v8bf lo = *(const v8bf*)(row + halfSel * 8);
    v8bf hi = *(const v8bf*)(row + 16 + halfSel * 8);
    v16bf r;
    #pragma unroll
    for (int i = 0; i < 8; i++) { r[i] = lo[i]; r[8 + i] = hi[i]; }
    return r;
}
__device__ __forceinline__ v16bf frag_b(const __bf16* row, int halfSel) {
    v8bf lo = *(const v8bf*)(row + halfSel * 8);
    v8bf hi = *(const v8bf*)(row + 16 + halfSel * 8);
    v16bf r;
    #pragma unroll
    for (int i = 0; i < 8; i++) { r[i] = lo[i]; r[8 + i] = hi[i]; }
    return r;
}

// ---------------------------------------------------------------------------
// C[M,N] = act(alpha * A[M,K] @ B[N,K]^T + bias)
// Outputs (any subset): C (f32), Cb (bf16, same layout), CbT (bf16 per-head
// transposed V: [b, h, 64, TT]). Batched per blockIdx.z: b=z/H, h=z%H.
// grid = (N/64, M/128, B*H), block = 256.
// ---------------------------------------------------------------------------
struct GemmP {
    const __bf16* A; const __bf16* B;
    float* C; __bf16* Cb; __bf16* CbT;
    const float* bias;
    int K, lda, ldb, ldc;
    long long sAb, sAh, sBb, sBh, sCb, sCh;
    int H; float alpha; int actGelu; int TT;
};

__global__ __launch_bounds__(256) void gemm_bf16(GemmP P)
{
    __shared__ __attribute__((aligned(16))) __bf16 sA[2][128][LDSP];
    __shared__ __attribute__((aligned(16))) __bf16 sB[2][64][LDSP];

    const int bz = blockIdx.z;
    const int bb = bz / P.H, hh = bz % P.H;
    const __bf16* A = P.A + (size_t)bb * P.sAb + (size_t)hh * P.sAh;
    const __bf16* B = P.B + (size_t)bb * P.sBb + (size_t)hh * P.sBh;

    const int rowBase = blockIdx.y * 128;
    const int colBase = blockIdx.x * 64;

    const int tid = threadIdx.x, lane = tid & 31, wave = tid >> 5;
    const int waveM = wave >> 1;       // 0..3 : 32-row strip
    const int waveN = wave & 1;        // 0..1 : 32-col strip
    const int halfSel = lane >> 4, lidx = lane & 15;

    // staging: 16B chunks; A tile 128x32 = 512 chunks (2/thread), B 64x32 = 256
    const int crow = tid >> 2;             // 0..63
    const int cseg = (tid & 3) * 8;        // 0,8,16,24 (bf16 elems)

    const unsigned baseA = (unsigned)(uintptr_t)&sA[0][0][0];
    const unsigned baseB = (unsigned)(uintptr_t)&sB[0][0][0];
    const unsigned ldsA0 = baseA + (unsigned)((crow * LDSP + cseg) * 2);
    const unsigned ldsA1 = baseA + (unsigned)(((crow + 64) * LDSP + cseg) * 2);
    const unsigned ldsB0 = baseB + (unsigned)((crow * LDSP + cseg) * 2);
    const unsigned bufA = 128 * LDSP * 2;
    const unsigned bufB = 64 * LDSP * 2;

    const __bf16* gA0 = A + (size_t)(rowBase + crow) * P.lda + cseg;
    const __bf16* gA1 = A + (size_t)(rowBase + crow + 64) * P.lda + cseg;
    const __bf16* gB0 = B + (size_t)(colBase + crow) * P.ldb + cseg;

    v8f acc[2][2] = {};
    const int nk = P.K / 32;

    // prologue: tile 0 into buffer 0
    async_b128(ldsA0, gA0);
    async_b128(ldsA1, gA1);
    async_b128(ldsB0, gB0);

    for (int kt = 0; kt < nk; ++kt) {
        const int cur = kt & 1;
        wait_async0();          // this wave's copies for tile kt have landed
        __syncthreads();        // whole block sees tile kt in LDS

        if (kt + 1 < nk) {      // overlap: issue tile kt+1 into other buffer
            const int k0 = (kt + 1) * 32;
            const unsigned bo = (cur ^ 1) ? 1u : 0u;
            async_b128(ldsA0 + bo * bufA, gA0 + k0);
            async_b128(ldsA1 + bo * bufA, gA1 + k0);
            async_b128(ldsB0 + bo * bufB, gB0 + k0);
        }

        const int mr0 = waveM * 32 + lidx, mr1 = mr0 + 16;
        const int nc0 = waveN * 32 + lidx, nc1 = nc0 + 16;
        v16bf a0 = frag_a(&sA[cur][mr0][0], halfSel);
        v16bf a1 = frag_a(&sA[cur][mr1][0], halfSel);
        v16bf b0 = frag_b(&sB[cur][nc0][0], halfSel);
        v16bf b1 = frag_b(&sB[cur][nc1][0], halfSel);

        acc[0][0] = __builtin_amdgcn_wmma_f32_16x16x32_bf16(false, a0, false, b0, (short)0, acc[0][0], false, false);
        acc[0][1] = __builtin_amdgcn_wmma_f32_16x16x32_bf16(false, a0, false, b1, (short)0, acc[0][1], false, false);
        acc[1][0] = __builtin_amdgcn_wmma_f32_16x16x32_bf16(false, a1, false, b0, (short)0, acc[1][0], false, false);
        acc[1][1] = __builtin_amdgcn_wmma_f32_16x16x32_bf16(false, a1, false, b1, (short)0, acc[1][1], false, false);
    }

    float*  C  = P.C  ? P.C  + (size_t)bb * P.sCb + (size_t)hh * P.sCh : nullptr;
    __bf16* Cb = P.Cb ? P.Cb + (size_t)bb * P.sCb + (size_t)hh * P.sCh : nullptr;

    #pragma unroll
    for (int im = 0; im < 2; im++) {
        const int mB = rowBase + waveM * 32 + im * 16 + halfSel * 8;
        #pragma unroll
        for (int jn = 0; jn < 2; jn++) {
            const int nB = colBase + waveN * 32 + jn * 16 + lidx;
            const float bs = P.bias ? P.bias[nB] : 0.f;
            #pragma unroll
            for (int r = 0; r < 8; r++) {
                float v = acc[im][jn][r] * P.alpha + bs;
                if (P.actGelu) v = 0.5f * v * (1.f + erff(v * 0.70710678118654752f));
                const int m = mB + r;
                if (C)  C[(size_t)m * P.ldc + nB] = v;
                if (Cb) Cb[(size_t)m * P.ldc + nB] = cvt_bf16(v);
                if (P.CbT) {   // per-head transposed V: [b,16,64,TT]
                    const int b2 = m / P.TT, t = m % P.TT;
                    const int h2 = nB >> 6, nl = nB & 63;
                    P.CbT[(((size_t)b2 * 16 + h2) * 64 + nl) * P.TT + t] = cvt_bf16(v);
                }
            }
        }
    }
}

// ---------------------------------------------------------------------------
// softmax: f32 logits in -> bf16 probabilities out. One block per row.
// ---------------------------------------------------------------------------
__global__ __launch_bounds__(256)
void softmax_bf16(const float* __restrict__ in, __bf16* __restrict__ out, int L)
{
    const float* row = in + (size_t)blockIdx.x * (size_t)L;
    __bf16* orow = out + (size_t)blockIdx.x * (size_t)L;
    __shared__ float sred[256];
    const int t = threadIdx.x;
    const int per = L / 256;          // <= 16
    float ev[16];

    float mx = -3.4e38f;
    for (int i = 0; i < per; i++) mx = fmaxf(mx, row[t + i * 256]);
    sred[t] = mx; __syncthreads();
    for (int s = 128; s > 0; s >>= 1) { if (t < s) sred[t] = fmaxf(sred[t], sred[t + s]); __syncthreads(); }
    mx = sred[0]; __syncthreads();

    float sum = 0.f;
    for (int i = 0; i < per; i++) { ev[i] = __expf(row[t + i * 256] - mx); sum += ev[i]; }
    sred[t] = sum; __syncthreads();
    for (int s = 128; s > 0; s >>= 1) { if (t < s) sred[t] += sred[t + s]; __syncthreads(); }
    const float inv = 1.f / sred[0];
    for (int i = 0; i < per; i++) orow[t + i * 256] = cvt_bf16(ev[i] * inv);
}

// ---------------------------------------------------------------------------
// outF = LayerNorm(a+b)*g+beta (f32); optional bf16 copy outB. Block per row.
// ---------------------------------------------------------------------------
__global__ __launch_bounds__(256)
void add_layernorm(const float* __restrict__ a, const float* __restrict__ b,
                   const float* __restrict__ g, const float* __restrict__ be,
                   float* __restrict__ outF, __bf16* __restrict__ outB, int D)
{
    const size_t base = (size_t)blockIdx.x * (size_t)D;
    __shared__ float sred[256];
    const int t = threadIdx.x;
    const int per = D / 256;
    float vals[8];

    float sum = 0.f;
    for (int i = 0; i < per; i++) {
        float v = a[base + t + i * 256] + b[base + t + i * 256];
        vals[i] = v; sum += v;
    }
    sred[t] = sum; __syncthreads();
    for (int s = 128; s > 0; s >>= 1) { if (t < s) sred[t] += sred[t + s]; __syncthreads(); }
    const float mean = sred[0] / (float)D; __syncthreads();

    float vs = 0.f;
    for (int i = 0; i < per; i++) { float d = vals[i] - mean; vs += d * d; }
    sred[t] = vs; __syncthreads();
    for (int s = 128; s > 0; s >>= 1) { if (t < s) sred[t] += sred[t + s]; __syncthreads(); }
    const float rstd = rsqrtf(sred[0] / (float)D + 1e-5f);

    for (int i = 0; i < per; i++) {
        int c = t + i * 256;
        float o = (vals[i] - mean) * rstd * g[c] + be[c];
        outF[base + c] = o;
        if (outB) outB[base + c] = cvt_bf16(o);
    }
}

// ---------------------------------------------------------------------------
// conversion helpers
// ---------------------------------------------------------------------------
__global__ __launch_bounds__(256)
void cvt_f32_bf16_k(const float* __restrict__ in, __bf16* __restrict__ out, int n)
{
    int i = blockIdx.x * 256 + threadIdx.x;
    if (i < n) out[i] = cvt_bf16(in[i]);
}
// in[K,N] f32 -> out[N,K] bf16
__global__ __launch_bounds__(256)
void cvtT_f32_bf16_k(const float* __restrict__ in, __bf16* __restrict__ out, int K, int N)
{
    int i = blockIdx.x * 256 + threadIdx.x;
    if (i < N * K) {
        int n = i / K, k = i % K;
        out[i] = cvt_bf16(in[(size_t)k * N + n]);
    }
}

// ---------------------------------------------------------------------------
// host side
// ---------------------------------------------------------------------------
static inline void run_gemm(hipStream_t s, const GemmP& P, int M, int N, int BH)
{
    dim3 g(N / 64, M / 128, BH), b(256);
    gemm_bf16<<<g, b, 0, s>>>(P);
}

extern "C" void kernel_launch(void* const* d_in, const int* in_sizes, int n_in,
                              void* d_out, int out_size, void* d_ws, size_t ws_size,
                              hipStream_t stream)
{
    (void)in_sizes; (void)n_in; (void)out_size; (void)ws_size;

    const int Bc = 4, NX = 4096, NP = 256, D = 1024, H = 16, MLPD = 4096;
    const int MX = Bc * NX, MP = Bc * NP;
    const float SCALE = 0.125f;

    const float* x    = (const float*)d_in[0];
    const float* p    = (const float*)d_in[1];
    const float* pq_w = (const float*)d_in[2];
    const float* pk_w = (const float*)d_in[3];
    const float* pv_w = (const float*)d_in[4];
    const float* po_w = (const float*)d_in[5];
    const float* po_b = (const float*)d_in[6];
    const float* xq_w = (const float*)d_in[7];
    const float* xk_w = (const float*)d_in[8];
    const float* xv_w = (const float*)d_in[9];
    const float* xo_w = (const float*)d_in[10];
    const float* xo_b = (const float*)d_in[11];
    const float* pn_g = (const float*)d_in[12];
    const float* pn_b = (const float*)d_in[13];
    const float* xn_g = (const float*)d_in[14];
    const float* xn_b = (const float*)d_in[15];
    const float* w1   = (const float*)d_in[16];
    const float* b1   = (const float*)d_in[17];
    const float* w2   = (const float*)d_in[18];
    const float* b2   = (const float*)d_in[19];
    const float* fn_g = (const float*)d_in[20];
    const float* fn_b = (const float*)d_in[21];

    float* out_x = (float*)d_out;
    float* out_p = out_x + (size_t)MX * D;

    // ---- workspace carve-up ----
    size_t off = 0;
    char* wsc = (char*)d_ws;
    auto alloc = [&](size_t bytes) { void* r = wsc + off; off += (bytes + 255) & ~(size_t)255; return r; };

    float*  dotsF = (float*)alloc((size_t)64 * NP * NX * 4);        // 268 MB (scores; later x_att/x_ff/h1)
    __bf16* attnB = (__bf16*)alloc((size_t)64 * NP * NX * 2);       // 134 MB
    __bf16* xb    = (__bf16*)alloc((size_t)MX * D * 2);
    __bf16* pb    = (__bf16*)alloc((size_t)MP * D * 2);
    __bf16* wpqT  = (__bf16*)alloc((size_t)D * D * 2);
    __bf16* wpkT  = (__bf16*)alloc((size_t)D * D * 2);
    __bf16* wpvT  = (__bf16*)alloc((size_t)D * D * 2);
    __bf16* wpoT  = (__bf16*)alloc((size_t)D * D * 2);
    __bf16* wxqT  = (__bf16*)alloc((size_t)D * D * 2);
    __bf16* wxkT  = (__bf16*)alloc((size_t)D * D * 2);
    __bf16* wxvT  = (__bf16*)alloc((size_t)D * D * 2);
    __bf16* wxoT  = (__bf16*)alloc((size_t)D * D * 2);
    __bf16* w1T   = (__bf16*)alloc((size_t)D * MLPD * 2);           // [4096,1024]
    __bf16* w2T   = (__bf16*)alloc((size_t)MLPD * D * 2);           // [1024,4096]
    __bf16* q_pb  = (__bf16*)alloc((size_t)MP * D * 2);
    __bf16* k_xb  = (__bf16*)alloc((size_t)MX * D * 2);             // later q_x
    __bf16* v_xT  = (__bf16*)alloc((size_t)MX * D * 2);             // later o_x
    __bf16* o_pb  = (__bf16*)alloc((size_t)MP * D * 2);
    float*  p_attF= (float*) alloc((size_t)MP * D * 4);
    __bf16* p_attB= (__bf16*)alloc((size_t)MP * D * 2);
    __bf16* k_pb  = (__bf16*)alloc((size_t)MP * D * 2);
    __bf16* v_pT  = (__bf16*)alloc((size_t)MP * D * 2);
    float*  x1F   = (float*) alloc((size_t)MX * D * 4);
    __bf16* x1B   = (__bf16*)alloc((size_t)MX * D * 2);
    // sub-regions of dotsF after scores are consumed:
    float*  x_attF = dotsF;                                          // 64 MB
    float*  x_ffF  = dotsF + (size_t)MX * D;                         // 64 MB
    __bf16* h1B    = (__bf16*)((char*)dotsF + (size_t)128 * 1024 * 1024); // 134 MB

    auto cvt  = [&](const float* i, __bf16* o, size_t n) {
        cvt_f32_bf16_k<<<(unsigned)((n + 255) / 256), 256, 0, stream>>>(i, o, (int)n);
    };
    auto cvtT = [&](const float* i, __bf16* o, int K, int N) {
        cvtT_f32_bf16_k<<<(unsigned)(((size_t)K * N + 255) / 256), 256, 0, stream>>>(i, o, K, N);
    };

    cvt(x, xb, (size_t)MX * D);
    cvt(p, pb, (size_t)MP * D);
    cvtT(pq_w, wpqT, D, D);  cvtT(pk_w, wpkT, D, D);
    cvtT(pv_w, wpvT, D, D);  cvtT(po_w, wpoT, D, D);
    cvtT(xq_w, wxqT, D, D);  cvtT(xk_w, wxkT, D, D);
    cvtT(xv_w, wxvT, D, D);  cvtT(xo_w, wxoT, D, D);
    cvtT(w1, w1T, D, MLPD);  cvtT(w2, w2T, MLPD, D);

    GemmP P{};
    auto clear = [&]() { P = GemmP{}; P.H = 1; P.alpha = 1.f; P.TT = 1; };

    // ============ Phase A: p attends to x ============
    clear(); P.A = pb; P.B = wpqT; P.Cb = q_pb; P.K = D; P.lda = P.ldb = P.ldc = D;
    run_gemm(stream, P, MP, D, 1);
    clear(); P.A = xb; P.B = wpkT; P.Cb = k_xb; P.K = D; P.lda = P.ldb = P.ldc = D;
    run_gemm(stream, P, MX, D, 1);
    clear(); P.A = xb; P.B = wpvT; P.CbT = v_xT; P.TT = NX; P.K = D; P.lda = P.ldb = P.ldc = D;
    run_gemm(stream, P, MX, D, 1);

    // dots[b,h] = SCALE * q_p @ k_x^T -> f32 [256,4096]
    clear(); P.A = q_pb; P.B = k_xb; P.C = dotsF; P.K = 64;
    P.lda = D; P.ldb = D; P.ldc = NX;
    P.sAb = (long long)NP * D; P.sAh = 64;
    P.sBb = (long long)NX * D; P.sBh = 64;
    P.sCb = (long long)H * NP * NX; P.sCh = (long long)NP * NX;
    P.H = H; P.alpha = SCALE;
    run_gemm(stream, P, NP, NX, Bc * H);

    softmax_bf16<<<Bc * H * NP, 256, 0, stream>>>(dotsF, attnB, NX);

    // o_p[b,h] = attn @ v -> bf16 [256,64]
    clear(); P.A = attnB; P.B = v_xT; P.Cb = o_pb; P.K = NX;
    P.lda = NX; P.ldb = NX; P.ldc = D;
    P.sAb = (long long)H * NP * NX; P.sAh = (long long)NP * NX;
    P.sBb = (long long)H * 64 * NX; P.sBh = (long long)64 * NX;
    P.sCb = (long long)NP * D; P.sCh = 64;
    P.H = H;
    run_gemm(stream, P, NP, 64, Bc * H);

    clear(); P.A = o_pb; P.B = wpoT; P.C = p_attF; P.Cb = p_attB; P.bias = po_b;
    P.K = D; P.lda = P.ldb = P.ldc = D;
    run_gemm(stream, P, MP, D, 1);

    add_layernorm<<<MP, 256, 0, stream>>>(p_attF, p, pn_g, pn_b, out_p, nullptr, D);

    // ============ Phase B: x attends to p_att ============
    __bf16* q_xb = k_xb;   // reuse
    __bf16* o_xb = v_xT;   // reuse
    clear(); P.A = xb; P.B = wxqT; P.Cb = q_xb; P.K = D; P.lda = P.ldb = P.ldc = D;
    run_gemm(stream, P, MX, D, 1);
    clear(); P.A = p_attB; P.B = wxkT; P.Cb = k_pb; P.K = D; P.lda = P.ldb = P.ldc = D;
    run_gemm(stream, P, MP, D, 1);
    clear(); P.A = p_attB; P.B = wxvT; P.CbT = v_pT; P.TT = NP; P.K = D; P.lda = P.ldb = P.ldc = D;
    run_gemm(stream, P, MP, D, 1);

    // dots2[b,h] = SCALE * q_x @ k_p^T -> f32 [4096,256]
    clear(); P.A = q_xb; P.B = k_pb; P.C = dotsF; P.K = 64;
    P.lda = D; P.ldb = D; P.ldc = NP;
    P.sAb = (long long)NX * D; P.sAh = 64;
    P.sBb = (long long)NP * D; P.sBh = 64;
    P.sCb = (long long)H * NX * NP; P.sCh = (long long)NX * NP;
    P.H = H; P.alpha = SCALE;
    run_gemm(stream, P, NX, NP, Bc * H);

    softmax_bf16<<<Bc * H * NX, 256, 0, stream>>>(dotsF, attnB, NP);

    // o_x[b,h] = attn2 @ v_p -> bf16 [4096,64]
    clear(); P.A = attnB; P.B = v_pT; P.Cb = o_xb; P.K = NP;
    P.lda = NP; P.ldb = NP; P.ldc = D;
    P.sAb = (long long)H * NX * NP; P.sAh = (long long)NX * NP;
    P.sBb = (long long)H * 64 * NP; P.sBh = (long long)64 * NP;
    P.sCb = (long long)NX * D; P.sCh = 64;
    P.H = H;
    run_gemm(stream, P, NX, 64, Bc * H);

    clear(); P.A = o_xb; P.B = wxoT; P.C = x_attF; P.bias = xo_b;
    P.K = D; P.lda = P.ldb = P.ldc = D;
    run_gemm(stream, P, MX, D, 1);

    add_layernorm<<<MX, 256, 0, stream>>>(x_attF, x, xn_g, xn_b, x1F, x1B, D);

    // ============ MLP ============
    clear(); P.A = x1B; P.B = w1T; P.Cb = h1B; P.bias = b1; P.actGelu = 1;
    P.K = D; P.lda = D; P.ldb = D; P.ldc = MLPD;
    run_gemm(stream, P, MX, MLPD, 1);

    clear(); P.A = h1B; P.B = w2T; P.C = x_ffF; P.bias = b2;
    P.K = MLPD; P.lda = MLPD; P.ldb = MLPD; P.ldc = D;
    run_gemm(stream, P, MX, D, 1);

    add_layernorm<<<MX, 256, 0, stream>>>(x_ffF, x1F, fn_g, fn_b, out_x, nullptr, D);
}
